// ALIGNModule_72842645340579
// MI455X (gfx1250) — matrix operand order repaired and checked by reading, not compiled, not run on hardware
//
#include <hip/hip_runtime.h>
#include <hip/hip_bf16.h>

typedef __bf16 bf16_t;
typedef __attribute__((ext_vector_type(4)))  bf16_t v4bf;
typedef __attribute__((ext_vector_type(8)))  bf16_t v8bf;
typedef __attribute__((ext_vector_type(16))) bf16_t v16bf;
typedef __attribute__((ext_vector_type(8)))  float  v8f;

#define LN_EPS 1e-5f
#define NEG_BIG -1e30f

// ---------- helpers ----------
static __device__ __forceinline__ v8f zero8() {
  v8f z;
#pragma unroll
  for (int i = 0; i < 8; ++i) z[i] = 0.f;
  return z;
}

// Assemble a 16-element bf16 fragment from two 16-byte chunks (b128 loads).
static __device__ __forceinline__ v16bf load_ab(const bf16_t* p0, const bf16_t* p1) {
  v8bf lo = *(const v8bf*)p0;
  v8bf hi = *(const v8bf*)p1;
  v16bf r;
#pragma unroll
  for (int i = 0; i < 8; ++i) { r[i] = lo[i]; r[i + 8] = hi[i]; }
  return r;
}

static __device__ __forceinline__ v8f wmma_bf16(v16bf a, v16bf b, v8f c) {
  return __builtin_amdgcn_wmma_f32_16x16x32_bf16(false, a, false, b, (short)0, c, false, false);
}

// ============================================================================
// Kernel 1: pixel-shuffle + GEMM1 (+bias) -> Ftmp f32 (2048 x 2048)
// ============================================================================
__global__ __launch_bounds__(256) void k_gemm1(const float* __restrict__ X,
                                               const float* __restrict__ W1,
                                               const float* __restrict__ B1,
                                               float* __restrict__ F) {
  __shared__ __align__(16) bf16_t As[128 * 40];
  __shared__ __align__(16) bf16_t Bs[64 * 40];

  const int t = threadIdx.x;
  const int lane = t & 31, w = t >> 5;
  const int ln = lane & 15, h = lane >> 4;
  const int bm = blockIdx.y, bn = blockIdx.x;
  const int wm = w >> 1, wn = w & 1;

  const int ar = t >> 1, ah = t & 1;
  const int m = bm * 128 + ar;
  const int pb = m >> 8, L = m & 255, i2 = L >> 4, j2 = L & 15;
  const int br = t >> 2, bq = t & 3;
  const long wrow = (long)(bn * 64 + br) * 4608;

  v8f acc[2][2];
#pragma unroll
  for (int i = 0; i < 2; ++i)
#pragma unroll
    for (int j = 0; j < 2; ++j) acc[i][j] = zero8();

  for (int k0 = 0; k0 < 4608; k0 += 32) {
    __syncthreads();
    {
      const int a = k0 / 2304, e = k0 % 2304, c = e / 1152, d0 = e % 1152;
      const float* src =
          X + (((long)pb * 32 + 2 * i2 + a) * 32 + (2 * j2 + c)) * 1152 + d0 + 16 * ah;
#pragma unroll
      for (int q = 0; q < 4; ++q) {
        float4 v = *(const float4*)(src + 4 * q);
        bf16_t* dst = As + ar * 40 + 16 * ah + 4 * q;
        dst[0] = (bf16_t)v.x; dst[1] = (bf16_t)v.y;
        dst[2] = (bf16_t)v.z; dst[3] = (bf16_t)v.w;
      }
      const float* srcB = W1 + wrow + k0 + 4 * bq;
#pragma unroll
      for (int q = 0; q < 2; ++q) {
        float4 v = *(const float4*)(srcB + 16 * q);
        bf16_t* dst = Bs + br * 40 + 16 * q + 4 * bq;
        dst[0] = (bf16_t)v.x; dst[1] = (bf16_t)v.y;
        dst[2] = (bf16_t)v.z; dst[3] = (bf16_t)v.w;
      }
    }
    __syncthreads();
#pragma unroll
    for (int im = 0; im < 2; ++im) {
      const bf16_t* ap = As + (32 * wm + 16 * im + ln) * 40 + 16 * h;
      v16bf af = load_ab(ap, ap + 8);
#pragma unroll
      for (int in = 0; in < 2; ++in) {
        const bf16_t* bp = Bs + (32 * wn + 16 * in + ln) * 40 + 16 * h;
        v16bf bfr = load_ab(bp, bp + 8);
        acc[im][in] = wmma_bf16(af, bfr, acc[im][in]);
      }
    }
  }
#pragma unroll
  for (int im = 0; im < 2; ++im)
#pragma unroll
    for (int in = 0; in < 2; ++in) {
      const int ncol = bn * 64 + 32 * wn + 16 * in + ln;
      const float bias = B1[ncol];
      const int mrow0 = bm * 128 + 32 * wm + 16 * im + 8 * h;
#pragma unroll
      for (int v = 0; v < 8; ++v)
        F[(long)(mrow0 + v) * 2048 + ncol] = acc[im][in][v] + bias;
    }
}

// ============================================================================
// Kernel 2: layernorm over last dim (2048), write bf16 F
// ============================================================================
__global__ __launch_bounds__(256) void k_layernorm(const float* __restrict__ F,
                                                   bf16_t* __restrict__ Fb) {
  __shared__ float rs[8], rs2[8];
  __shared__ float smu, srsig;
  const int m = blockIdx.x, t = threadIdx.x;
  const float* row = F + (long)m * 2048;
  float4 x0 = ((const float4*)row)[t * 2];
  float4 x1 = ((const float4*)row)[t * 2 + 1];
  float s = x0.x + x0.y + x0.z + x0.w + x1.x + x1.y + x1.z + x1.w;
  float s2 = x0.x * x0.x + x0.y * x0.y + x0.z * x0.z + x0.w * x0.w +
             x1.x * x1.x + x1.y * x1.y + x1.z * x1.z + x1.w * x1.w;
#pragma unroll
  for (int off = 16; off > 0; off >>= 1) {
    s += __shfl_xor(s, off);
    s2 += __shfl_xor(s2, off);
  }
  if ((t & 31) == 0) { rs[t >> 5] = s; rs2[t >> 5] = s2; }
  __syncthreads();
  if (t == 0) {
    float S = 0.f, S2 = 0.f;
#pragma unroll
    for (int i = 0; i < 8; ++i) { S += rs[i]; S2 += rs2[i]; }
    const float mu = S * (1.f / 2048.f);
    const float var = S2 * (1.f / 2048.f) - mu * mu;
    smu = mu;
    srsig = rsqrtf(var + LN_EPS);
  }
  __syncthreads();
  const float mu = smu, r = srsig;
  v8bf o;
  o[0] = (bf16_t)((x0.x - mu) * r); o[1] = (bf16_t)((x0.y - mu) * r);
  o[2] = (bf16_t)((x0.z - mu) * r); o[3] = (bf16_t)((x0.w - mu) * r);
  o[4] = (bf16_t)((x1.x - mu) * r); o[5] = (bf16_t)((x1.y - mu) * r);
  o[6] = (bf16_t)((x1.z - mu) * r); o[7] = (bf16_t)((x1.w - mu) * r);
  *(v8bf*)(Fb + (long)m * 2048 + t * 8) = o;
}

// ============================================================================
// Pack kernels (packed path): W2 -> bf16 [v][k]; E -> bf16 transposed [n][v]
// ============================================================================
__global__ __launch_bounds__(256) void k_pack_w2(const float* __restrict__ W2,
                                                 bf16_t* __restrict__ Wb) {
  const long i = ((long)blockIdx.x * 256 + threadIdx.x) * 8;
  float4 a = *(const float4*)(W2 + i);
  float4 b = *(const float4*)(W2 + i + 4);
  v8bf o;
  o[0] = (bf16_t)a.x; o[1] = (bf16_t)a.y; o[2] = (bf16_t)a.z; o[3] = (bf16_t)a.w;
  o[4] = (bf16_t)b.x; o[5] = (bf16_t)b.y; o[6] = (bf16_t)b.z; o[7] = (bf16_t)b.w;
  *(v8bf*)(Wb + i) = o;
}

__global__ __launch_bounds__(256) void k_pack_et(const float* __restrict__ E,
                                                 bf16_t* __restrict__ Et) {
  __shared__ __align__(16) bf16_t tileT[32][36];  // [n][v], padded
  const int t = threadIdx.x;
  const int v0 = blockIdx.x * 32, n0 = blockIdx.y * 32;
  {
    const int r = t >> 3, c4 = t & 7;  // r: v row, c4: n quarter-of-4
    float4 x = *(const float4*)(E + (long)(v0 + r) * 2048 + n0 + 4 * c4);
    tileT[4 * c4 + 0][r] = (bf16_t)x.x;
    tileT[4 * c4 + 1][r] = (bf16_t)x.y;
    tileT[4 * c4 + 2][r] = (bf16_t)x.z;
    tileT[4 * c4 + 3][r] = (bf16_t)x.w;
  }
  __syncthreads();
  {
    const int c = t >> 3, q = t & 7;  // c: n row, q: v quarter-of-4
    *(v4bf*)(Et + (long)(n0 + c) * 32000 + v0 + 4 * q) = *(const v4bf*)&tileT[c][4 * q];
  }
}

// ============================================================================
// Kernel 3a (PACKED): fused GEMM2 + online softmax + GEMM3, V step = 64,
//   B-fragments straight from L2-resident bf16 (Wb row-major, Et transposed).
//   No staging LDS; only deterministic S-reduction + P.
// ============================================================================
#define VTOT 32000
#define VSPLIT 2
#define VPART (VTOT / VSPLIT)

__global__ __launch_bounds__(256) void k_flash_packed(
    const bf16_t* __restrict__ Fb, const bf16_t* __restrict__ Wb,
    const bf16_t* __restrict__ Et, float* __restrict__ Pbuf,
    float* __restrict__ mstat, float* __restrict__ lstat) {
  __shared__ __align__(16) float Sred8[8 * 16 * 64];  // 32 KB
  __shared__ __align__(16) bf16_t Ps[16 * 72];        // P, row stride 72
  __shared__ float facs[16];
  __shared__ float mrow[16], lrow[16];

  const int t = threadIdx.x, lane = t & 31, w = t >> 5;
  const int ln = lane & 15, h = lane >> 4;
  const int mb = blockIdx.x >> 1;
  const int vp = blockIdx.x & 1;
  const int m0 = mb * 16;
  const int vbeg = vp * VPART, vend = vbeg + VPART;

  v8f acc[16];
#pragma unroll
  for (int j = 0; j < 16; ++j) acc[j] = zero8();
  if (t < 16) { mrow[t] = NEG_BIG; lrow[t] = 0.f; }

  for (int v0 = vbeg; v0 < vend; v0 += 64) {
    if (v0 + 64 < vend) {  // stream-ahead hints for next step's tiles
      __builtin_prefetch(Wb + (long)(v0 + 64 + 16 * h + ln) * 2048 + 256 * w, 0, 1);
      __builtin_prefetch(Wb + (long)(v0 + 96 + 16 * h + ln) * 2048 + 256 * w, 0, 1);
      __builtin_prefetch(Et + (long)(256 * w + 16 * h + ln) * 32000 + v0 + 64, 0, 1);
      __builtin_prefetch(Et + (long)(256 * w + 128 + 16 * h + ln) * 32000 + v0 + 64, 0, 1);
    }
    // ---- partial S = F x W2^T over this wave's K slice (256 wide) ----
    v8f s[4];
#pragma unroll
    for (int q = 0; q < 4; ++q) s[q] = zero8();
#pragma unroll
    for (int kk = 0; kk < 8; ++kk) {
      const int kg = 256 * w + 32 * kk;
      const bf16_t* ap = Fb + (long)(m0 + ln) * 2048 + kg + 16 * h;
      v16bf af = load_ab(ap, ap + 8);
#pragma unroll
      for (int q = 0; q < 4; ++q) {
        const bf16_t* bp = Wb + (long)(v0 + 16 * q + ln) * 2048 + kg + 16 * h;
        s[q] = wmma_bf16(af, load_ab(bp, bp + 8), s[q]);
      }
    }
    {
      float* myS = Sred8 + w * 1024;
#pragma unroll
      for (int q = 0; q < 4; ++q)
#pragma unroll
        for (int v = 0; v < 8; ++v)
          myS[(8 * h + v) * 64 + 16 * q + ln] = s[q][v];
    }
    __syncthreads();
    // ---- deterministic cross-wave reduce (1024 entries, 4/thread) ----
#pragma unroll
    for (int e = t; e < 1024; e += 256) {
      float sum = 0.f;
#pragma unroll
      for (int ww = 0; ww < 8; ++ww) sum += Sred8[ww * 1024 + e];
      Sred8[e] = sum;
    }
    __syncthreads();
    // ---- online softmax update (16 threads, 64 cols, two-pass) ----
    if (t < 16) {
      const float mold = mrow[t];
      float mx = mold;
#pragma unroll
      for (int j = 0; j < 64; ++j) mx = fmaxf(mx, Sred8[t * 64 + j]);
      const float fac = __expf(mold - mx);
      float sum = 0.f;
#pragma unroll
      for (int j = 0; j < 64; ++j) {
        const float p = __expf(Sred8[t * 64 + j] - mx);
        sum += p;
        Ps[t * 72 + j] = (bf16_t)p;
      }
      lrow[t] = lrow[t] * fac + sum;
      mrow[t] = mx;
      facs[t] = fac;
    }
    __syncthreads();
    // ---- rescale acc, then acc += P x E via Et (direct global B-frags) ----
    float fv[8];
#pragma unroll
    for (int v = 0; v < 8; ++v) fv[v] = facs[8 * h + v];
#pragma unroll
    for (int j = 0; j < 16; ++j)
#pragma unroll
      for (int v = 0; v < 8; ++v) acc[j][v] *= fv[v];
    const bf16_t* pp = Ps + ln * 72 + 16 * h;
    v16bf pf0 = load_ab(pp, pp + 8);        // K = v 0..31
    v16bf pf1 = load_ab(pp + 32, pp + 40);  // K = v 32..63
#pragma unroll
    for (int jj = 0; jj < 16; ++jj) {
      const int n = 256 * w + 16 * jj;
      const bf16_t* bp0 = Et + (long)(n + ln) * 32000 + v0 + 16 * h;
      acc[jj] = wmma_bf16(pf0, load_ab(bp0, bp0 + 8), acc[jj]);
      const bf16_t* bp1 = bp0 + 32;
      acc[jj] = wmma_bf16(pf1, load_ab(bp1, bp1 + 8), acc[jj]);
    }
  }
  __syncthreads();
  if (t < 16) {
    mstat[vp * 2048 + m0 + t] = mrow[t];
    lstat[vp * 2048 + m0 + t] = lrow[t];
  }
  float* PO = Pbuf + (long)vp * 2048 * 2048;
#pragma unroll
  for (int jj = 0; jj < 16; ++jj) {
    const int n = 256 * w + 16 * jj;
#pragma unroll
    for (int v = 0; v < 8; ++v)
      PO[(long)(m0 + 8 * h + v) * 2048 + n + ln] = acc[jj][v];
  }
}

// ============================================================================
// Kernel 3b (STAGED fallback, small ws): round-1 version, V step = 32
// ============================================================================
__global__ __launch_bounds__(256) void k_flash_staged(
    const bf16_t* __restrict__ Fb, const float* __restrict__ W2,
    const float* __restrict__ E, float* __restrict__ Pbuf,
    float* __restrict__ mstat, float* __restrict__ lstat) {
  __shared__ __align__(16) bf16_t stage[512 * 40];
  __shared__ __align__(16) float Sred8[8 * 16 * 32];
  __shared__ __align__(16) bf16_t Ps[16 * 40];
  __shared__ float facs[16];
  __shared__ float mrow[16], lrow[16];

  const int t = threadIdx.x, lane = t & 31, w = t >> 5;
  const int ln = lane & 15, h = lane >> 4;
  const int mb = blockIdx.x >> 1;
  const int vp = blockIdx.x & 1;
  const int m0 = mb * 16;
  const int vbeg = vp * VPART, vend = vbeg + VPART;

  v8f acc[16];
#pragma unroll
  for (int j = 0; j < 16; ++j) acc[j] = zero8();
  if (t < 16) { mrow[t] = NEG_BIG; lrow[t] = 0.f; }

  for (int v0 = vbeg; v0 < vend; v0 += 32) {
    v8f s0 = zero8(), s1 = zero8();
    for (int kq = 0; kq < 4; ++kq) {
      __syncthreads();
      for (int i = t; i < 32 * 128; i += 256) {
        const int vv = i >> 7, c4 = i & 127;
        float4 x = *(const float4*)(W2 + (long)(v0 + vv) * 2048 + 512 * kq + 4 * c4);
        bf16_t* d = stage + vv * 520 + 4 * c4;
        d[0] = (bf16_t)x.x; d[1] = (bf16_t)x.y; d[2] = (bf16_t)x.z; d[3] = (bf16_t)x.w;
      }
      __syncthreads();
#pragma unroll
      for (int kk = 0; kk < 2; ++kk) {
        const int kg = 512 * kq + 64 * w + 32 * kk;
        const bf16_t* ap = Fb + (long)(m0 + ln) * 2048 + kg + 16 * h;
        v16bf af = load_ab(ap, ap + 8);
        const bf16_t* bp0 = stage + ln * 520 + (64 * w + 32 * kk) + 16 * h;
        s0 = wmma_bf16(af, load_ab(bp0, bp0 + 8), s0);
        const bf16_t* bp1 = stage + (16 + ln) * 520 + (64 * w + 32 * kk) + 16 * h;
        s1 = wmma_bf16(af, load_ab(bp1, bp1 + 8), s1);
      }
    }
    {
      float* myS = Sred8 + w * 512;
#pragma unroll
      for (int v = 0; v < 8; ++v) {
        myS[(8 * h + v) * 32 + ln] = s0[v];
        myS[(8 * h + v) * 32 + 16 + ln] = s1[v];
      }
    }
    __syncthreads();
#pragma unroll
    for (int e = t; e < 512; e += 256) {
      float sum = 0.f;
#pragma unroll
      for (int ww = 0; ww < 8; ++ww) sum += Sred8[ww * 512 + e];
      Sred8[e] = sum;
    }
    __syncthreads();
    if (t < 16) {
      const float mold = mrow[t];
      float mx = mold;
#pragma unroll
      for (int j = 0; j < 32; ++j) mx = fmaxf(mx, Sred8[t * 32 + j]);
      const float fac = __expf(mold - mx);
      float sum = 0.f;
#pragma unroll
      for (int j = 0; j < 32; ++j) {
        const float p = __expf(Sred8[t * 32 + j] - mx);
        sum += p;
        Ps[t * 40 + j] = (bf16_t)p;
      }
      lrow[t] = lrow[t] * fac + sum;
      mrow[t] = mx;
      facs[t] = fac;
    }
    __syncthreads();
    float fv[8];
#pragma unroll
    for (int v = 0; v < 8; ++v) fv[v] = facs[8 * h + v];
#pragma unroll
    for (int j = 0; j < 16; ++j)
#pragma unroll
      for (int v = 0; v < 8; ++v) acc[j][v] *= fv[v];
    const bf16_t* pp = Ps + ln * 40 + 16 * h;
    v16bf pf = load_ab(pp, pp + 8);
    for (int nq = 0; nq < 4; ++nq) {
      __syncthreads();
      for (int i = t; i < 32 * 128; i += 256) {
        const int vv = i >> 7, c4 = i & 127;
        float4 x = *(const float4*)(E + (long)(v0 + vv) * 2048 + 512 * nq + 4 * c4);
        stage[(4 * c4 + 0) * 40 + vv] = (bf16_t)x.x;
        stage[(4 * c4 + 1) * 40 + vv] = (bf16_t)x.y;
        stage[(4 * c4 + 2) * 40 + vv] = (bf16_t)x.z;
        stage[(4 * c4 + 3) * 40 + vv] = (bf16_t)x.w;
      }
      __syncthreads();
#pragma unroll
      for (int jq = 0; jq < 4; ++jq) {
        const int jj = 4 * nq + jq;
        const int nl = 128 * jq + 16 * w;
        const bf16_t* bp = stage + (nl + ln) * 40 + 16 * h;
        acc[jj] = wmma_bf16(pf, load_ab(bp, bp + 8), acc[jj]);
      }
    }
  }
  __syncthreads();
  if (t < 16) {
    mstat[vp * 2048 + m0 + t] = mrow[t];
    lstat[vp * 2048 + m0 + t] = lrow[t];
  }
  float* PO = Pbuf + (long)vp * 2048 * 2048;
#pragma unroll
  for (int jj = 0; jj < 16; ++jj) {
    const int n = 128 * jj + 16 * w + ln;
#pragma unroll
    for (int v = 0; v < 8; ++v)
      PO[(long)(m0 + 8 * h + v) * 2048 + n] = acc[jj][v];
  }
}

// ============================================================================
// Kernel 4: log-sum-exp combine of the two V partitions
// ============================================================================
__global__ __launch_bounds__(256) void k_combine(const float* __restrict__ Pbuf,
                                                 const float* __restrict__ mstat,
                                                 const float* __restrict__ lstat,
                                                 float* __restrict__ out) {
  const int m = blockIdx.x, t = threadIdx.x;
  const float m1 = mstat[m], m2 = mstat[2048 + m];
  const float l1 = lstat[m], l2 = lstat[2048 + m];
  const float M = fmaxf(m1, m2);
  const float a1 = __expf(m1 - M), a2 = __expf(m2 - M);
  const float inv = 1.f / (a1 * l1 + a2 * l2);
  const float c1 = a1 * inv, c2 = a2 * inv;
  const float4* p1 = (const float4*)(Pbuf + (long)m * 2048) + t * 2;
  const float4* p2 = (const float4*)(Pbuf + (long)2048 * 2048 + (long)m * 2048) + t * 2;
  float4* o = (float4*)(out + (long)m * 2048) + t * 2;
#pragma unroll
  for (int q = 0; q < 2; ++q) {
    float4 x = p1[q], y = p2[q], r;
    r.x = c1 * x.x + c2 * y.x;
    r.y = c1 * x.y + c2 * y.y;
    r.z = c1 * x.z + c2 * y.z;
    r.w = c1 * x.w + c2 * y.w;
    o[q] = r;
  }
}

// ============================================================================
extern "C" void kernel_launch(void* const* d_in, const int* in_sizes, int n_in,
                              void* d_out, int out_size, void* d_ws, size_t ws_size,
                              hipStream_t stream) {
  const float* X  = (const float*)d_in[0];
  const float* E  = (const float*)d_in[1];
  const float* W1 = (const float*)d_in[2];
  const float* B1 = (const float*)d_in[3];
  const float* W2 = (const float*)d_in[4];
  float* out = (float*)d_out;

  char* ws = (char*)d_ws;
  float*  Ftmp  = (float*)(ws + 0);                  // 16 MB
  bf16_t* Fb    = (bf16_t*)(ws + 16777216);          // 8 MB
  float*  Pbuf  = (float*)(ws + 25165824);           // 32 MB
  float*  mstat = (float*)(ws + 58720256);
  float*  lstat = (float*)(ws + 58736640);
  bf16_t* Wb    = (bf16_t*)(ws + 58753024);          // 131 MB (packed path)
  bf16_t* Et    = (bf16_t*)(ws + 58753024 + 131072000ull);  // 131 MB (packed path)
  const size_t WS_PACKED_NEED = 58753024ull + 2ull * 131072000ull;  // ~306 MiB

  dim3 g1(32, 16);
  k_gemm1<<<g1, 256, 0, stream>>>(X, W1, B1, Ftmp);
  k_layernorm<<<2048, 256, 0, stream>>>(Ftmp, Fb);

  if (ws_size >= WS_PACKED_NEED) {
    k_pack_w2<<<32000, 256, 0, stream>>>(W2, Wb);
    dim3 gt(1000, 64);  // (V/32, N/32)
    k_pack_et<<<gt, 256, 0, stream>>>(E, Et);
    k_flash_packed<<<128 * VSPLIT, 256, 0, stream>>>(Fb, Wb, Et, Pbuf, mstat, lstat);
  } else {
    k_flash_staged<<<128 * VSPLIT, 256, 0, stream>>>(Fb, W2, E, Pbuf, mstat, lstat);
  }
  k_combine<<<2048, 256, 0, stream>>>(Pbuf, mstat, lstat, out);
}
